// AutoregressiveLSTMClassifier_1468878815277
// MI455X (gfx1250) — matrix operand order repaired and checked by reading, not compiled, not run on hardware
//
#include <hip/hip_runtime.h>
#include <math.h>

#define B_ 64
#define S_ 512
#define D_ 1024
#define H_ 1024
#define E_ 128
#define V_ 128
#define DE_ (D_ + E_)   // 1152

typedef __attribute__((ext_vector_type(16))) _Float16 v16h;
typedef __attribute__((ext_vector_type(8)))  _Float16 v8h;
typedef __attribute__((ext_vector_type(8)))  float    v8f;

// ---- WMMA fragment loaders (contiguous k order, same for A and B, wave32) ----
// A 16x32 f16 tile: lane l holds row m=(l&15); elements 0..15 = K kb..kb+15,
// kb = k0 + 16*(l>>4).
static __device__ inline v16h load_a_from_f16(const _Float16* __restrict__ p) {
    v8h lo = *(const v8h*)(p);
    v8h hi = *(const v8h*)(p + 8);
    v16h a;
#pragma unroll
    for (int i = 0; i < 8; ++i) { a[i] = lo[i]; a[8 + i] = hi[i]; }
    return a;
}

// B 32x16 f16 tile of W^T: lane l holds weight row n=(l&15), 16 contiguous K
// starting at k0 + 16*(l>>4). 32B-aligned single load.
static __device__ inline v16h load_b(const _Float16* __restrict__ p) {
    return *(const v16h*)(p);
}

static __device__ inline float sigmoidf_(float x) {
    return 1.0f / (1.0f + __expf(-x));
}

#define WMMA_F16(a, b, c) \
    __builtin_amdgcn_wmma_f32_16x16x32_f16(false, (a), false, (b), (short)0, (c), false, false)

// =====================================================================
// Kernel 1: gates GEMM + LSTM pointwise + next-step x conversion.
// grid = 64 WGs x 256 threads.  Wave -> (gate, n-tile, m-half): 2 M-tiles
// with 2 independent accumulators; B fragment reused from registers.
// WG = 2 tile-groups x 4 gates -> i/f/g/o combined in LDS.
// =====================================================================
__global__ __launch_bounds__(256) void lstm_gates_step(
    const float*    __restrict__ x,       // slot_hidden [B,S,D] fp32
    const int*      __restrict__ attn,    // [B,S]
    const float*    __restrict__ bsum,    // [4H] = b_ih + b_hh
    const _Float16* __restrict__ Wih,     // [4H, DE] f16
    const _Float16* __restrict__ Whh,     // [4H, H] f16
    const _Float16* __restrict__ xh,      // [B, D] f16  x_t (read)
    _Float16*       __restrict__ xh_next, // [B, D] f16  x_{t+1} (write)
    const _Float16* __restrict__ prev,    // [B, E] f16
    const _Float16* __restrict__ hprev,   // [B, H] f16 (masked h, read)
    _Float16*       __restrict__ hnew,    // [B, H] f16 (unmasked, for logits)
    _Float16*       __restrict__ hnext,   // [B, H] f16 (masked, write)
    float*          __restrict__ c,       // [B, H] fp32 (in/out)
    int t)
{
    __shared__ float lds_g[2][4][2][256];   // [grp][gate][mtile][elem] = 16 KB

    const int tid  = threadIdx.x;
    const int wave = tid >> 5;
    const int lane = tid & 31;
    const int l15  = lane & 15;
    const int lhi  = lane >> 4;

    // ---- stage x_{t+1} -> f16 (overlaps with GEMM below) ----
    if (t + 1 < S_) {
        for (int i = blockIdx.x * 256 + tid; i < B_ * D_; i += 64 * 256) {
            int b = i >> 10;            // / D_
            int d = i & (D_ - 1);
            xh_next[i] = (_Float16)x[((size_t)b * S_ + (t + 1)) * D_ + d];
        }
    }

    const int grp    = wave >> 2;              // 0..1
    const int gate   = wave & 3;               // i,f,g,o
    const int gid    = blockIdx.x * 2 + grp;   // 0..127
    const int m_half = gid >> 6;               // 0..1  (batch rows 32*m_half..)
    const int n_tile = gid & 63;               // 0..63
    const int brow   = gate * H_ + n_tile * 16 + l15;
    const int arow0  = m_half * 32 + l15;      // first M-tile's row

    v8f acc[2] = {};

    // ---- xcat @ Wih^T : K = 1152 (x part, then prev part) ----
    const _Float16* wb = Wih + (size_t)brow * DE_;
#pragma unroll 2
    for (int k0 = 0; k0 < D_; k0 += 32) {
        v16h b = load_b(wb + k0 + 16 * lhi);
#pragma unroll
        for (int mt = 0; mt < 2; ++mt) {
            v16h a = load_a_from_f16(xh + (size_t)(arow0 + mt * 16) * D_ + k0 + 16 * lhi);
            acc[mt] = WMMA_F16(a, b, acc[mt]);
        }
    }
#pragma unroll
    for (int k0 = 0; k0 < E_; k0 += 32) {
        v16h b = load_b(wb + D_ + k0 + 16 * lhi);
#pragma unroll
        for (int mt = 0; mt < 2; ++mt) {
            v16h a = load_a_from_f16(prev + (size_t)(arow0 + mt * 16) * E_ + k0 + 16 * lhi);
            acc[mt] = WMMA_F16(a, b, acc[mt]);
        }
    }
    // ---- h @ Whh^T : K = 1024 ----
    const _Float16* wb2 = Whh + (size_t)brow * H_;
#pragma unroll 2
    for (int k0 = 0; k0 < H_; k0 += 32) {
        v16h b = load_b(wb2 + k0 + 16 * lhi);
#pragma unroll
        for (int mt = 0; mt < 2; ++mt) {
            v16h a = load_a_from_f16(hprev + (size_t)(arow0 + mt * 16) * H_ + k0 + 16 * lhi);
            acc[mt] = WMMA_F16(a, b, acc[mt]);
        }
    }

    // D layout: element r, lane l -> (m = r + 8*(l>>4), n = l&15)
#pragma unroll
    for (int mt = 0; mt < 2; ++mt)
#pragma unroll
        for (int r = 0; r < 8; ++r)
            lds_g[grp][gate][mt][32 * r + lane] = acc[mt][r];
    __syncthreads();

    // ---- pointwise LSTM: 1024 elements in this WG, 4 per thread ----
#pragma unroll
    for (int e2 = 0; e2 < 4; ++e2) {
        int e   = tid + e2 * 256;
        int eg  = e >> 9;           // group 0..1
        int w   = e & 511;
        int ml  = w >> 4;           // 0..31 local batch row
        int n   = w & 15;
        int mtl = ml >> 4;          // 0..1
        int mm  = ml & 15;
        int li  = 32 * (mm & 7) + n + 16 * (mm >> 3);

        int g_gid  = blockIdx.x * 2 + eg;
        int g_mh   = g_gid >> 6;
        int g_nt   = g_gid & 63;
        int bm     = g_mh * 32 + ml;             // batch index
        int bn     = g_nt * 16 + n;              // hidden index

        float gi = lds_g[eg][0][mtl][li] + bsum[0 * H_ + bn];
        float gf = lds_g[eg][1][mtl][li] + bsum[1 * H_ + bn];
        float gg = lds_g[eg][2][mtl][li] + bsum[2 * H_ + bn];
        float go = lds_g[eg][3][mtl][li] + bsum[3 * H_ + bn];

        float ig = sigmoidf_(gi);
        float fg = sigmoidf_(gf);
        float gt = tanhf(gg);
        float og = sigmoidf_(go);

        size_t idx  = (size_t)bm * H_ + bn;
        float  cold = c[idx];
        float  cnew = fg * cold + ig * gt;
        float  hn   = og * tanhf(cnew);

        bool msk   = attn[bm * S_ + t] > 0;
        hnew[idx]  = (_Float16)hn;                      // pre-mask h for logits
        hnext[idx] = msk ? (_Float16)hn : hprev[idx];   // carried (masked) h
        if (msk) c[idx] = cnew;
    }
}

// =====================================================================
// Kernel 2: logits GEMM [64x128] + argmax + log-softmax + emb feedback.
// grid = 4 WGs (one per M-tile) x 256 threads; wave w -> n-tile w.
// Cooperative row reduction: 16 threads per batch row.
// =====================================================================
__global__ __launch_bounds__(256) void logits_step(
    const _Float16* __restrict__ hnew,   // [B,H] f16 (pre-mask)
    const _Float16* __restrict__ Wlin,   // [V,H] f16
    const float*    __restrict__ b_lin,  // [V]
    const _Float16* __restrict__ emb_h,  // [V,E] f16
    const int*      __restrict__ attn,   // [B,S]
    _Float16*       __restrict__ prev,   // [B,E] f16 (in/out)
    float*          __restrict__ out,    // [B,S,V]
    int t)
{
    __shared__ float lds_l[16 * V_];     // this m-tile's 16 logit rows (8 KB)
    __shared__ float red_mx[16][16];
    __shared__ int   red_am[16][16];
    __shared__ float red_s[16][16];
    __shared__ float row_mx[16];
    __shared__ float row_lse[16];
    __shared__ int   row_am[16];

    const int tid  = threadIdx.x;
    const int wave = tid >> 5;           // n-tile 0..7
    const int lane = tid & 31;
    const int l15  = lane & 15;
    const int lhi  = lane >> 4;
    const int m_tile = blockIdx.x;       // 0..3

    v8f acc = {};
    const _Float16* arow = hnew + (size_t)(m_tile * 16 + l15) * H_;
    const _Float16* wrow = Wlin + (size_t)(wave * 16 + l15) * H_;
#pragma unroll 4
    for (int k0 = 0; k0 < H_; k0 += 32) {
        v16h a = load_a_from_f16(arow + k0 + 16 * lhi);
        v16h b = load_b(wrow + k0 + 16 * lhi);
        acc = WMMA_F16(a, b, acc);
    }
    {
        int   n  = wave * 16 + l15;
        float bl = b_lin[n];
#pragma unroll
        for (int r = 0; r < 8; ++r) {
            int mloc = r + 8 * lhi;      // 0..15
            lds_l[mloc * V_ + n] = acc[r] + bl;
        }
    }
    __syncthreads();

    // ---- cooperative argmax / log-sum-exp: 16 threads per row ----
    const int r  = tid >> 4;             // row 0..15
    const int cx = tid & 15;             // chunk 0..15 (8 logits each)
    const float* rowp = &lds_l[r * V_ + cx * 8];

    float mx = rowp[0];
    int   am = cx * 8;
#pragma unroll
    for (int j = 1; j < 8; ++j) {
        float xv = rowp[j];
        if (xv > mx) { mx = xv; am = cx * 8 + j; }
    }
    red_mx[r][cx] = mx;
    red_am[r][cx] = am;
    __syncthreads();
    if (cx == 0) {
        float bm = red_mx[r][0];
        int   ba = red_am[r][0];
        for (int i = 1; i < 16; ++i) {
            float v = red_mx[r][i];
            if (v > bm) { bm = v; ba = red_am[r][i]; }   // first-max overall
        }
        row_mx[r] = bm;
        row_am[r] = ba;
    }
    __syncthreads();
    {
        float rm = row_mx[r];
        float s  = 0.0f;
#pragma unroll
        for (int j = 0; j < 8; ++j) s += __expf(rowp[j] - rm);
        red_s[r][cx] = s;
    }
    __syncthreads();
    if (cx == 0) {
        float s = 0.0f;
        for (int i = 0; i < 16; ++i) s += red_s[r][i];
        row_lse[r] = row_mx[r] + __logf(s);
    }
    __syncthreads();

    {
        int    b    = m_tile * 16 + r;
        bool   msk  = attn[b * S_ + t] > 0;
        float* orow = out + ((size_t)b * S_ + t) * V_ + cx * 8;
        if (msk) {
            float lse = row_lse[r];
#pragma unroll
            for (int j = 0; j < 8; ++j) orow[j] = rowp[j] - lse;
            const _Float16* esrc = emb_h + (size_t)row_am[r] * E_ + cx * 8;
            _Float16*       edst = prev + (size_t)b * E_ + cx * 8;
#pragma unroll
            for (int j = 0; j < 8; ++j) edst[j] = esrc[j];
        } else {
            const float lc = -4.852030263919617f;   // log_softmax of zero row
#pragma unroll
            for (int j = 0; j < 8; ++j) orow[j] = lc;
        }
    }
}

// =====================================================================
// One-time per-launch init: f16 weight conversion, bias precombine,
// state zeroing, x[:,0,:] staging.
// =====================================================================
__global__ void init_convert(
    const float* __restrict__ Wih,  const float* __restrict__ Whh,
    const float* __restrict__ Wlin, const float* __restrict__ emb,
    const float* __restrict__ init_t, const float* __restrict__ x,
    const float* __restrict__ b_ih, const float* __restrict__ b_hh,
    _Float16* __restrict__ Wih_h,  _Float16* __restrict__ Whh_h,
    _Float16* __restrict__ Wlin_h, _Float16* __restrict__ emb_h,
    _Float16* __restrict__ prev_h, _Float16* __restrict__ hbuf0,
    _Float16* __restrict__ hbuf1,  float* __restrict__ c,
    _Float16* __restrict__ xh0,    float* __restrict__ bsum)
{
    size_t i0 = (size_t)blockIdx.x * blockDim.x + threadIdx.x;
    size_t st = (size_t)gridDim.x * blockDim.x;
    for (size_t i = i0; i < (size_t)4 * H_ * DE_; i += st) Wih_h[i]  = (_Float16)Wih[i];
    for (size_t i = i0; i < (size_t)4 * H_ * H_;  i += st) Whh_h[i]  = (_Float16)Whh[i];
    for (size_t i = i0; i < (size_t)V_ * H_;      i += st) Wlin_h[i] = (_Float16)Wlin[i];
    for (size_t i = i0; i < (size_t)V_ * E_;      i += st) emb_h[i]  = (_Float16)emb[i];
    for (size_t i = i0; i < (size_t)B_ * E_;      i += st) prev_h[i] = (_Float16)init_t[i % E_];
    for (size_t i = i0; i < (size_t)4 * H_;       i += st) bsum[i]   = b_ih[i] + b_hh[i];
    for (size_t i = i0; i < (size_t)B_ * H_;      i += st) {
        hbuf0[i] = (_Float16)0.0f;
        hbuf1[i] = (_Float16)0.0f;
        c[i]     = 0.0f;
        size_t b = i >> 10, d = i & (D_ - 1);
        xh0[i] = (_Float16)x[(b * S_) * D_ + d];   // x[:,0,:] staging
    }
}

// =====================================================================
extern "C" void kernel_launch(void* const* d_in, const int* in_sizes, int n_in,
                              void* d_out, int out_size, void* d_ws, size_t ws_size,
                              hipStream_t stream) {
    const float* x      = (const float*)d_in[0];  // slot_hidden
    const int*   attn   = (const int*)  d_in[1];  // attention_mask
    const float* Wih    = (const float*)d_in[2];
    const float* Whh    = (const float*)d_in[3];
    const float* b_ih   = (const float*)d_in[4];
    const float* b_hh   = (const float*)d_in[5];
    const float* Wlin   = (const float*)d_in[6];
    const float* b_lin  = (const float*)d_in[7];
    const float* emb    = (const float*)d_in[8];
    const float* init_t = (const float*)d_in[9];
    float* out = (float*)d_out;

    char* ws = (char*)d_ws;
    _Float16* Wih_h  = (_Float16*)(ws + 0);          //  9,437,184 B
    _Float16* Whh_h  = (_Float16*)(ws + 9437184);    //  8,388,608 B
    _Float16* Wlin_h = (_Float16*)(ws + 17825792);   //    262,144 B
    _Float16* emb_h  = (_Float16*)(ws + 18087936);   //     32,768 B
    _Float16* prev_h = (_Float16*)(ws + 18120704);   //     16,384 B
    _Float16* hnew_h = (_Float16*)(ws + 18137088);   //    131,072 B
    _Float16* hbuf0  = (_Float16*)(ws + 18268160);   //    131,072 B
    _Float16* hbuf1  = (_Float16*)(ws + 18399232);   //    131,072 B
    float*    c_f    = (float*)   (ws + 18530304);   //    262,144 B
    _Float16* xh0    = (_Float16*)(ws + 18792448);   //    131,072 B
    _Float16* xh1    = (_Float16*)(ws + 18923520);   //    131,072 B
    float*    bsum   = (float*)   (ws + 19054592);   //     16,384 B
    // total ~19.1 MB of workspace

    init_convert<<<2048, 256, 0, stream>>>(Wih, Whh, Wlin, emb, init_t, x,
                                           b_ih, b_hh,
                                           Wih_h, Whh_h, Wlin_h, emb_h,
                                           prev_h, hbuf0, hbuf1, c_f, xh0, bsum);

    for (int t = 0; t < S_; ++t) {
        _Float16* hprev = (t & 1) ? hbuf1 : hbuf0;
        _Float16* hnext = (t & 1) ? hbuf0 : hbuf1;
        _Float16* xhc   = (t & 1) ? xh1 : xh0;
        _Float16* xhn   = (t & 1) ? xh0 : xh1;
        lstm_gates_step<<<64, 256, 0, stream>>>(x, attn, bsum,
                                                Wih_h, Whh_h, xhc, xhn, prev_h,
                                                hprev, hnew_h, hnext, c_f, t);
        logits_step<<<4, 256, 0, stream>>>(hnew_h, Wlin_h, b_lin, emb_h,
                                           attn, prev_h, out, t);
    }
}
